// ScaledDotProductAttention_73443940761810
// MI455X (gfx1250) — compile-verified, matrix operand order checked by reading, not run on hardware
//
#include <hip/hip_runtime.h>
#include <hip/hip_bf16.h>

// ---------------------------------------------------------------------------
// Fused per-position attention for MI455X (gfx1250, wave32, WMMA).
//
//   q_t = query @ Wq.T ; k_t = key @ Wk.T ; v_t = value @ Wv.T
//   scores[b,d,i] = <q_t[b,i,:], k_t[d,i,:]> * 1/sqrt(F)
//   attn = softmax_d(scores) ; out[b,i,:] = sum_d attn[b,d,i] * v_t[d,i,:]
//
// Everything is independent per position i -> fuse the whole chain per i.
// bf16 WMMA (16x16x32, f32 accum) makes the kernel HBM-bound (~23us floor).
//
// Round 3's runtime per-element address select (transpose ternary) made the
// compiler precompute+spill 128 LDS addresses.  Round 4: wave-uniform
// if(w==2) around the store loop -> compile-time strides, immediate-offset
// ds stores, no cndmask/spill; V path packs pairs into ds_store_b32.
// ---------------------------------------------------------------------------

typedef __attribute__((ext_vector_type(16))) __bf16 v16bf;
typedef __attribute__((ext_vector_type(8)))  __bf16 v8bf;
typedef __attribute__((ext_vector_type(8)))  float  v8f;
typedef __attribute__((ext_vector_type(4)))  float  f32x4;

#define BATCH 32
#define SEQ   8192
#define FEAT  128

__device__ __forceinline__ v16bf cat8(v8bf lo, v8bf hi) {
  return __builtin_shufflevector(lo, hi, 0,1,2,3,4,5,6,7,8,9,10,11,12,13,14,15);
}

__device__ __forceinline__ v8f wmma_bf16(v16bf a, v16bf b, v8f c) {
  // v_wmma_f32_16x16x32_bf16  (neg_a, A, neg_b, B, c_mod, C, reuse_a, reuse_b)
  return __builtin_amdgcn_wmma_f32_16x16x32_bf16(false, a, false, b, (short)0, c,
                                                 false, false);
}

// A-fragment (16x32 bf16) from a row-major fp32 row.  ISA layout: lane r
// (r=lane&15, hi=lane>>4) holds row M=r, elements j=0..15 -> K = hi*8 + j
// for j<8 and K = 16 + hi*8 + (j-8) for j>=8.  So per lane: two contiguous
// runs of 8 fp32 at p and p+16.
__device__ __forceinline__ v16bf load_a_f32(const float* p) {
  f32x4 a0 = *(const f32x4*)(p);
  f32x4 a1 = *(const f32x4*)(p + 4);
  f32x4 b0 = *(const f32x4*)(p + 16);
  f32x4 b1 = *(const f32x4*)(p + 20);
  v16bf r;
#pragma unroll
  for (int j = 0; j < 4; ++j) {
    r[j]      = (__bf16)a0[j];
    r[j + 4]  = (__bf16)a1[j];
    r[j + 8]  = (__bf16)b0[j];
    r[j + 12] = (__bf16)b1[j];
  }
  return r;
}

// Same fragment pattern from row-major bf16 (LDS): runs of 8 bf16 at p, p+16.
__device__ __forceinline__ v16bf load_frag_lds(const __bf16* p) {
  v8bf lo = *(const v8bf*)(p);
  v8bf hi = *(const v8bf*)(p + 16);
  return cat8(lo, hi);
}

// Pack two fp32 accumulator values into one b32 of two bf16 and store to LDS.
__device__ __forceinline__ void store_pk_bf16(__bf16* p, float lo, float hi) {
  union {
    struct { __bf16 a, b; } h;
    unsigned u;
  } pk;
  pk.h.a = (__bf16)lo;
  pk.h.b = (__bf16)hi;
  *(unsigned*)p = pk.u;
}

// ---------------------------------------------------------------------------
// Prep kernel: bf16-convert W^T into WMMA B-fragment-contiguous layout.
// Weight w (0..2), fragment (nt*4+kst), lane, element j -> 16 contiguous bf16
// per lane, 1KB per fragment, 32KB per weight, 96KB total in d_ws.
// B[k,n] = W[n,k] (nn.Linear x@W.T).  B fragment: lane column n = nt*16+r,
// element j at k = kst*32 + kk(j,hi) with the A-style K mapping.
// ---------------------------------------------------------------------------
__global__ void prep_weights(const float* __restrict__ Wq,
                             const float* __restrict__ Wk,
                             const float* __restrict__ Wv,
                             __bf16* __restrict__ wf) {
  int t = blockIdx.x * 256 + threadIdx.x;       // 3*32*32*16 = 49152 total
  if (t >= 3 * 32 * 32 * 16) return;
  int j    = t & 15;
  int lane = (t >> 4) & 31;
  int frag = (t >> 9) & 31;
  int w    = t >> 14;
  int r  = lane & 15, hi = lane >> 4;
  int nt = frag >> 2, kst = frag & 3;
  int p2 = j >> 1, q2 = j & 1;
  int kk = ((p2 < 4) ? (2 * p2) : (16 + 2 * (p2 - 4))) + q2 + hi * 8;
  int k  = kst * 32 + kk;
  int n  = nt * 16 + r;
  const float* W = (w == 0) ? Wq : (w == 1) ? Wk : Wv;
  wf[t] = (__bf16)W[n * FEAT + k];
}

// Per-wave LDS: Qb[32][128] + Kb[32][128] + Vt[128][32] + attn[32][32] bf16.
#define WAVE_LDS (4096 + 4096 + 4096 + 1024)   // 13312 bf16 = 26 KB

// min 4 waves per SIMD => <=256 VGPRs/wave.
__global__ void __launch_bounds__(128, 4)
fused_attention(const float* __restrict__ Q, const float* __restrict__ K,
                const float* __restrict__ V, const __bf16* __restrict__ WF,
                float* __restrict__ out) {
  extern __shared__ __bf16 smem[];
  const int lane = threadIdx.x & 31;
  const int wv   = threadIdx.x >> 5;          // wave in block (0..3)
  const int r = lane & 15, hi = lane >> 4;

  __bf16* ldsQ = smem + wv * WAVE_LDS;
  __bf16* ldsK = ldsQ + 4096;
  __bf16* ldsV = ldsK + 4096;                 // transposed: [f][d]
  __bf16* ldsA = ldsV + 4096;                 // attn: [b][d]

  const int gw = blockIdx.x * 4 + wv;         // global wave id, 0..2047
  const int NW = gridDim.x * 4;
  const float scale = 0.08838834764831845f;   // 1/sqrt(128)

  for (int i = gw; i < SEQ; i += NW) {
    // -------- prefetch next position's rows (global_prefetch_b8) ----------
    int ip = i + NW;
    if (ip < SEQ) {
#pragma unroll
      for (int mt = 0; mt < 2; ++mt) {
        long row = (long)(mt * 16 + r) * (long)(SEQ * FEAT) + (long)ip * FEAT;
        __builtin_prefetch(Q + row, 0, 1);
        __builtin_prefetch(Q + row + 64, 0, 1);
        __builtin_prefetch(K + row, 0, 1);
        __builtin_prefetch(K + row + 64, 0, 1);
        __builtin_prefetch(V + row, 0, 1);
        __builtin_prefetch(V + row + 64, 0, 1);
      }
    }

    // -------- three projection GEMMs (192 WMMAs), runtime w-loop ----------
    // #pragma unroll 1: keeps weight-fragment load addresses dependent on w
    // so nothing is loop-invariant in the position loop => no hoist/spill.
#pragma unroll 1
    for (int w = 0; w < 3; ++w) {
      const float*  xbase  = (w == 0) ? Q : (w == 1) ? K : V;
      const __bf16* wfp    = WF + w * 16384;
      __bf16*       ldsOut = (w == 0) ? ldsQ : (w == 1) ? ldsK : ldsV;
      xbase += (long)i * FEAT;

      v16bf A[2][4];
#pragma unroll
      for (int mt = 0; mt < 2; ++mt)
#pragma unroll
        for (int kst = 0; kst < 4; ++kst)
          A[mt][kst] = load_a_f32(xbase +
                                  (long)(mt * 16 + r) * (long)(SEQ * FEAT) +
                                  kst * 32 + hi * 8);
#pragma unroll
      for (int nt = 0; nt < 8; ++nt) {
        v16bf B[4];
#pragma unroll
        for (int kst = 0; kst < 4; ++kst) {
          const __bf16* p = wfp + (nt * 4 + kst) * 512 + lane * 16;
          B[kst] = cat8(*(const v8bf*)p, *(const v8bf*)(p + 8));
        }
#pragma unroll
        for (int mt = 0; mt < 2; ++mt) {
          v8f c = {};
#pragma unroll
          for (int kst = 0; kst < 4; ++kst) c = wmma_bf16(A[mt][kst], B[kst], c);

          if (w == 2) {
            // V path: transposed [f][d].  addr = N*32 + M; consecutive e are
            // adjacent bf16 -> pack pairs, 4x ds_store_b32.
            __bf16* base = ldsOut + (nt * 16 + r) * 32 + mt * 16 + hi * 8;
#pragma unroll
            for (int e2 = 0; e2 < 4; ++e2)
              store_pk_bf16(base + 2 * e2, c[2 * e2], c[2 * e2 + 1]);
          } else {
            // Q/K path: row-major [row][f].  Base VGPR + immediate e*FEAT.
            __bf16* base = ldsOut + (mt * 16 + hi * 8) * FEAT + nt * 16 + r;
#pragma unroll
            for (int e = 0; e < 8; ++e)
              base[e * FEAT] = (__bf16)c[e];
          }
        }
      }
    }

    // -------- scores S = Qb @ Kb^T (32x32, K=128) -------------------------
    v16bf qa[2][4], kb[2][4];
#pragma unroll
    for (int mt = 0; mt < 2; ++mt)
#pragma unroll
      for (int kst = 0; kst < 4; ++kst)
        qa[mt][kst] = load_frag_lds(ldsQ + (mt * 16 + r) * FEAT + kst * 32 + hi * 8);
#pragma unroll
    for (int nt = 0; nt < 2; ++nt)
#pragma unroll
      for (int kst = 0; kst < 4; ++kst)
        kb[nt][kst] = load_frag_lds(ldsK + (nt * 16 + r) * FEAT + kst * 32 + hi * 8);

    v8f sc[2][2];
#pragma unroll
    for (int mt = 0; mt < 2; ++mt)
#pragma unroll
      for (int nt = 0; nt < 2; ++nt) {
        v8f c = {};
#pragma unroll
        for (int kst = 0; kst < 4; ++kst) c = wmma_bf16(qa[mt][kst], kb[nt][kst], c);
        sc[mt][nt] = c;
      }

    // -------- softmax over d (across N: 16 lanes x 2 tiles), fp32 ---------
#pragma unroll
    for (int mt = 0; mt < 2; ++mt)
#pragma unroll
      for (int e = 0; e < 8; ++e) {
        float a0 = sc[mt][0][e] * scale;
        float a1 = sc[mt][1][e] * scale;
        float m = fmaxf(a0, a1);
        m = fmaxf(m, __shfl_xor(m, 1, 32));
        m = fmaxf(m, __shfl_xor(m, 2, 32));
        m = fmaxf(m, __shfl_xor(m, 4, 32));
        m = fmaxf(m, __shfl_xor(m, 8, 32));
        float e0 = __expf(a0 - m);
        float e1 = __expf(a1 - m);
        float s = e0 + e1;
        s += __shfl_xor(s, 1, 32);
        s += __shfl_xor(s, 2, 32);
        s += __shfl_xor(s, 4, 32);
        s += __shfl_xor(s, 8, 32);
        float inv = 1.0f / s;
        int b = mt * 16 + e + hi * 8;
        ldsA[b * 32 + r]      = (__bf16)(e0 * inv);
        ldsA[b * 32 + 16 + r] = (__bf16)(e1 * inv);
      }

    // -------- O = attn @ Vb (32x128, K=32) --------------------------------
    v16bf aa[2];
#pragma unroll
    for (int mt = 0; mt < 2; ++mt)
      aa[mt] = load_frag_lds(ldsA + (mt * 16 + r) * 32 + hi * 8);

#pragma unroll
    for (int nt = 0; nt < 8; ++nt) {
      v16bf vb = load_frag_lds(ldsV + (nt * 16 + r) * 32 + hi * 8);
#pragma unroll
      for (int mt = 0; mt < 2; ++mt) {
        v8f o = {};
        o = wmma_bf16(aa[mt], vb, o);
#pragma unroll
        for (int e = 0; e < 8; ++e) {
          int b = mt * 16 + e + hi * 8;
          int f = nt * 16 + r;
          out[((long)b * SEQ + i) * FEAT + f] = o[e];
        }
      }
    }
  }
}

extern "C" void kernel_launch(void* const* d_in, const int* in_sizes, int n_in,
                              void* d_out, int out_size, void* d_ws, size_t ws_size,
                              hipStream_t stream) {
  const float* q  = (const float*)d_in[0];
  const float* k  = (const float*)d_in[1];
  const float* v  = (const float*)d_in[2];
  const float* Wq = (const float*)d_in[3];
  const float* Wk = (const float*)d_in[4];
  const float* Wv = (const float*)d_in[5];
  __bf16* wf = (__bf16*)d_ws;                       // needs 96 KB of scratch
  float* out = (float*)d_out;

  prep_weights<<<192, 256, 0, stream>>>(Wq, Wk, Wv, wf);

  size_t shmem = (size_t)4 * WAVE_LDS * sizeof(__bf16);   // 104 KB
  fused_attention<<<512, 128, shmem, stream>>>(q, k, v, wf, out);
}
